// HybridBlock_88880053223663
// MI455X (gfx1250) — matrix operand order repaired and checked by reading, not compiled, not run on hardware
//
#include <hip/hip_runtime.h>
#include <hip/hip_bf16.h>
#include <math.h>

// ---------------------------------------------------------------------------
// HybridBlock for MI455X (gfx1250, wave32, WMMA, async-LDS).
//
//  x:(2,1024,32,32) ->
//   1) ln_kernel<true>  : LayerNorm(d,d) + dt = softplus(xn . w_dt + b_dt),
//                         log_a = dt * (-exp(log_A))          (per token)
//   2) ssm_scan_kernel  : h_t = exp(log_a_t)*h_{t-1} + xn_t ; ssm = h + x
//                         (exact rewrite of the causal exp(cs-cs) matmul)
//   3) ln_kernel<false> : LayerNorm of ssm -> fn
//   4) pack_w_kernel    : W1,W2 fp32 -> tiled bf16 hi/lo blocks (LDS image)
//   5) gemm_kernel<0>   : h = gelu(fn @ W1 + bb1)   (M=2048,K=1024,N=4096)
//   6) gemm_kernel<1>   : out = ssm + h @ W2 + bb2  (M=2048,K=4096,N=1024)
//
// GEMMs: v_wmma_f32_16x16x32_bf16 with bf16 hi/lo split (hi*hi+hi*lo+lo*hi,
// TF32x3-style fp32 emulation), fp32 accumulate. B tiles are pre-packed in
// global memory bit-identical to their LDS image, so staging is a pure
// linear copy -> GLOBAL_LOAD_ASYNC_TO_LDS_B128 (ASYNCcnt), with a plain
// uint4 copy fallback if the builtin is absent.
// ---------------------------------------------------------------------------

typedef __attribute__((ext_vector_type(16))) __bf16 v16bf;
typedef __attribute__((ext_vector_type(8)))  __bf16 v8bf;
typedef __attribute__((ext_vector_type(8)))  float  v8f;
typedef __attribute__((ext_vector_type(4)))  int    v4i;

#if defined(__has_builtin)
#  if __has_builtin(__builtin_amdgcn_global_load_async_to_lds_b128)
#    define USE_ASYNC_LDS 1
#  endif
#endif
#ifndef USE_ASYNC_LDS
#  define USE_ASYNC_LDS 0
#endif

// ---- bf16 helpers (round-to-nearest-even split) ---------------------------
__device__ __forceinline__ unsigned short f2bf(float f) {
    union { float f; unsigned u; } v; v.f = f;
    unsigned u = v.u;
    unsigned r = u + 0x7FFFu + ((u >> 16) & 1u);
    return (unsigned short)(r >> 16);
}
__device__ __forceinline__ float bf2f(unsigned short h) {
    union { unsigned u; float f; } v; v.u = ((unsigned)h) << 16;
    return v.f;
}

__device__ __forceinline__ float wave_sum32(float v) {
    #pragma unroll
    for (int o = 16; o > 0; o >>= 1) v += __shfl_xor(v, o, 32);
    return v;
}

#if USE_ASYNC_LDS
__device__ __forceinline__ void async_copy_b128(const void* g, void* l) {
    // builtin prototype (from hipcc diagnostic): param0 = AS1 pointer to v4i.
    typedef __attribute__((address_space(1))) v4i* gp_t;
    typedef __attribute__((address_space(3))) v4i* lp_t;
    __builtin_amdgcn_global_load_async_to_lds_b128(
        (gp_t)(__attribute__((address_space(1))) void*)((void*)g),
        (lp_t)(__attribute__((address_space(3))) void*)l,
        0, 0);
}
__device__ __forceinline__ void wait_async0() {
#  if __has_builtin(__builtin_amdgcn_s_wait_asynccnt)
    __builtin_amdgcn_s_wait_asynccnt(0);
#  else
    asm volatile("s_wait_asynccnt 0x0" ::: "memory");
#  endif
}
#endif

// ---------------------------------------------------------------------------
// Kernel 1/3: per-token LayerNorm over 1024 elements (one block per token,
// 256 threads = 8 wave32). Optionally computes dt-projection + log_a.
// ---------------------------------------------------------------------------
template <bool DO_DT>
__global__ __launch_bounds__(256) void ln_kernel(
    const float* __restrict__ in, const float* __restrict__ g,
    const float* __restrict__ bsh, const float* __restrict__ w_dt,
    const float* __restrict__ b_dt, const float* __restrict__ log_A,
    float* __restrict__ outn, float* __restrict__ loga)
{
    __shared__ float sh[1024];
    __shared__ float red[2][8];
    __shared__ float stats[2];

    const int tok = blockIdx.x;
    const int tid = threadIdx.x;
    const float4 v = ((const float4*)(in + (size_t)tok * 1024))[tid];

    float s = v.x + v.y + v.z + v.w;
    float q = v.x * v.x + v.y * v.y + v.z * v.z + v.w * v.w;
    s = wave_sum32(s);
    q = wave_sum32(q);
    const int lane = tid & 31, wid = tid >> 5;
    if (lane == 0) { red[0][wid] = s; red[1][wid] = q; }
    __syncthreads();
    if (tid == 0) {
        float ts = 0.f, tq = 0.f;
        #pragma unroll
        for (int i = 0; i < 8; ++i) { ts += red[0][i]; tq += red[1][i]; }
        const float m   = ts * (1.0f / 1024.0f);
        const float var = tq * (1.0f / 1024.0f) - m * m;
        stats[0] = m;
        stats[1] = rsqrtf(var + 1e-5f);
    }
    __syncthreads();
    const float m = stats[0], rstd = stats[1];

    const float4 gv = ((const float4*)g)[tid];
    const float4 bv = ((const float4*)bsh)[tid];
    float4 o;
    o.x = (v.x - m) * rstd * gv.x + bv.x;
    o.y = (v.y - m) * rstd * gv.y + bv.y;
    o.z = (v.z - m) * rstd * gv.z + bv.z;
    o.w = (v.w - m) * rstd * gv.w + bv.w;
    ((float4*)(outn + (size_t)tok * 1024))[tid] = o;
    ((float4*)sh)[tid] = o;
    __syncthreads();

    if (DO_DT && tid < 32) {
        const int r = tid;
        float acc = 0.f;
        #pragma unroll
        for (int c = 0; c < 32; ++c) acc += sh[r * 32 + c] * w_dt[c];
        acc += b_dt[0];
        const float sp = (acc > 20.f) ? acc : log1pf(expf(acc));
        loga[(size_t)tok * 32 + r] = sp * (-expf(log_A[r]));
    }
}

// ---------------------------------------------------------------------------
// Kernel 2: SSM causal mixing as a linear recurrence.
// Grid = 64 blocks (b in {0,1}, r in 0..31), 32 lanes over c.
// ssm[b,t,r,c] = h_t + x,  h_t = exp(log_a[b,t,r]) * h_{t-1} + xn[b,t,r,c]
// ---------------------------------------------------------------------------
__global__ void ssm_scan_kernel(const float* __restrict__ xn,
                                const float* __restrict__ loga,
                                const float* __restrict__ x,
                                float* __restrict__ ssm)
{
    const int b = blockIdx.x >> 5;
    const int r = blockIdx.x & 31;
    const int c = threadIdx.x;
    const size_t base = (size_t)b * 1024 * 1024 + (size_t)r * 32 + c;
    const float* la = loga + (size_t)b * 1024 * 32 + r;
    float h = 0.f;
    for (int t = 0; t < 1024; ++t) {
        const float a = expf(la[(size_t)t * 32]);
        const size_t idx = base + (size_t)t * 1024;
        h = fmaf(a, h, xn[idx]);
        ssm[idx] = h + x[idx];
    }
}

// ---------------------------------------------------------------------------
// Kernel 4: pack W (K x N fp32, row-major) into tiled bf16 hi/lo blocks that
// are bit-identical to the GEMM's LDS B-tile image (minus the LDS pad):
//   Wp[nt][ks][half][n:128][k:32]  (half 0 = hi, 1 = lo), 16 KB per (nt,ks).
// ---------------------------------------------------------------------------
__global__ __launch_bounds__(256) void pack_w_kernel(
    const float* __restrict__ W, unsigned short* __restrict__ Wp,
    int K, int N)
{
    const int nt = blockIdx.x;            // N/128
    const int ks = blockIdx.y;            // K/32
    const int n0 = nt * 128, k0 = ks * 32;
    unsigned short* dst = Wp + ((size_t)nt * (K >> 5) + ks) * 8192;
    const int tid = threadIdx.x;
    #pragma unroll
    for (int e = 0; e < 16; ++e) {
        const int idx = e * 256 + tid;    // 0..4095
        const int k = idx >> 7;           // 0..31
        const int n = idx & 127;          // 0..127
        const float f = W[(size_t)(k0 + k) * N + n0 + n];
        const unsigned short hi = f2bf(f);
        const unsigned short lo = f2bf(f - bf2f(hi));
        dst[n * 32 + k]        = hi;
        dst[4096 + n * 32 + k] = lo;
    }
}

// ---------------------------------------------------------------------------
// Kernels 5/6: fp32-A x packed-bf16-B GEMM via bf16x3 WMMA.
// Block: 256 threads (8 wave32), WG tile 128x128, wave tile 32x64.
// EPI=0: exact GELU -> out ;  EPI=1: out = resid + acc + bias.
// ---------------------------------------------------------------------------
#define LDT 40   // LDS row stride in bf16 elements (80B: 16B-aligned, padded)

__device__ __forceinline__ v16bf frag16(const unsigned short* p0,
                                        const unsigned short* p1) {
    const v8bf a0 = *(const v8bf*)p0;
    const v8bf a1 = *(const v8bf*)p1;
    return __builtin_shufflevector(a0, a1, 0, 1, 2, 3, 4, 5, 6, 7,
                                            8, 9, 10, 11, 12, 13, 14, 15);
}

template <int EPI>
__global__ __launch_bounds__(256) void gemm_kernel(
    const float* __restrict__ A, const unsigned short* __restrict__ Bp,
    const float* __restrict__ bias, const float* __restrict__ resid,
    float* __restrict__ out, int M, int N, int K)
{
    __shared__ __align__(16) unsigned short AsH[128 * LDT];
    __shared__ __align__(16) unsigned short AsL[128 * LDT];
    __shared__ __align__(16) unsigned short BtH[128 * LDT];
    __shared__ __align__(16) unsigned short BtL[128 * LDT];

    const int tid   = threadIdx.x;
    const int lane  = tid & 31, wid = tid >> 5;
    const int waveM = wid >> 1, waveN = wid & 1;     // 4 x 2 wave grid
    const int lm    = lane & 15, lhi = lane >> 4;
    const int m0    = blockIdx.y * 128;
    const int n0    = blockIdx.x * 128;

    v8f acc[2][4] = {};

    for (int k0 = 0; k0 < K; k0 += 32) {
        // ---- stage B: packed block is the LDS image -> pure linear copy.
        // 1024 chunks of 16B (hi tile then lo tile), 4 per thread.
        const unsigned short* bsrc =
            Bp + ((size_t)blockIdx.x * (K >> 5) + (k0 >> 5)) * 8192;
        #pragma unroll
        for (int it = 0; it < 4; ++it) {
            const int c    = tid + it * 256;       // 0..1023
            const int half = c >> 9;               // 0 = hi, 1 = lo
            const int cc   = c & 511;              // chunk within tile
            const int n    = cc >> 2;              // 0..127
            const int k8   = (cc & 3) << 3;        // 0,8,16,24
            const unsigned short* g = bsrc + half * 4096 + (size_t)cc * 8;
            unsigned short* l = (half ? BtL : BtH) + n * LDT + k8;
#if USE_ASYNC_LDS
            async_copy_b128(g, l);                 // global_load_async_to_lds_b128
#else
            *(uint4*)l = *(const uint4*)g;
#endif
        }

        // ---- stage A (128x32 fp32 -> hi/lo bf16 split in-flight)
        #pragma unroll
        for (int it = 0; it < 4; ++it) {
            const int fid = tid + it * 256;        // 0..1023
            const int r = fid >> 3;
            const int c = (fid & 7) << 2;
            const float4 av =
                *(const float4*)(A + (size_t)(m0 + r) * K + k0 + c);
            const float f[4] = {av.x, av.y, av.z, av.w};
            #pragma unroll
            for (int qq = 0; qq < 4; ++qq) {
                const unsigned short hi = f2bf(f[qq]);
                const unsigned short lo = f2bf(f[qq] - bf2f(hi));
                AsH[r * LDT + c + qq] = hi;
                AsL[r * LDT + c + qq] = lo;
            }
        }
#if USE_ASYNC_LDS
        wait_async0();                             // my async copies landed
#endif
        __syncthreads();                           // everyone's tiles visible

        // pull next K-slab of A toward L2 while we compute
        if (k0 + 32 < K) {
            __builtin_prefetch(A + (size_t)(m0 + (tid >> 1)) * K + k0 + 32, 0, 0);
        }

        // ---- A fragments: 16x32 bf16 layout (elems 0..7 -> K=8*lhi..,
        //      elems 8..15 -> K=16+8*lhi..) => two 16B LDS loads each.
        v16bf aH[2], aL[2];
        #pragma unroll
        for (int i = 0; i < 2; ++i) {
            const int row = waveM * 32 + i * 16 + lm;
            aH[i] = frag16(&AsH[row * LDT + 8 * lhi],
                           &AsH[row * LDT + 16 + 8 * lhi]);
            aL[i] = frag16(&AsL[row * LDT + 8 * lhi],
                           &AsL[row * LDT + 16 + 8 * lhi]);
        }

        // ---- B fragments (32x16): elems 0..7 -> K = 8*lhi.., elems 8..15 -> K = 16+8*lhi.., N = lane&15.
        #pragma unroll
        for (int j = 0; j < 4; ++j) {
            const int n = waveN * 64 + j * 16 + lm;
            const v16bf bH = frag16(&BtH[n * LDT + 8 * lhi],
                                    &BtH[n * LDT + 16 + 8 * lhi]);
            const v16bf bL = frag16(&BtL[n * LDT + 8 * lhi],
                                    &BtL[n * LDT + 16 + 8 * lhi]);
            #pragma unroll
            for (int i = 0; i < 2; ++i) {
                acc[i][j] = __builtin_amdgcn_wmma_f32_16x16x32_bf16(
                    false, aH[i], false, bH, (short)0, acc[i][j], false, false);
                acc[i][j] = __builtin_amdgcn_wmma_f32_16x16x32_bf16(
                    false, aL[i], false, bH, (short)0, acc[i][j], false, false);
                acc[i][j] = __builtin_amdgcn_wmma_f32_16x16x32_bf16(
                    false, aH[i], false, bL, (short)0, acc[i][j], false, false);
            }
        }
        __syncthreads();
    }

    // ---- epilogue: C/D layout = 8 VGPRs, m = r + 8*lhi, n = lane&15
    #pragma unroll
    for (int i = 0; i < 2; ++i) {
        #pragma unroll
        for (int j = 0; j < 4; ++j) {
            #pragma unroll
            for (int r = 0; r < 8; ++r) {
                const int gm = m0 + waveM * 32 + i * 16 + (r + 8 * lhi);
                const int gn = n0 + waveN * 64 + j * 16 + lm;
                const size_t idx = (size_t)gm * N + gn;
                float v = acc[i][j][r] + bias[gn];
                if (EPI == 0) {
                    v = 0.5f * v * (1.0f + erff(v * 0.70710678118654752f));
                } else {
                    v += resid[idx];
                }
                out[idx] = v;
            }
        }
    }
}

// ---------------------------------------------------------------------------
// Launch. Inputs (setup_inputs order):
//  0:x 1:log_A 2:w_dt 3:b_dt 4:g_ssm 5:b_ssm 6:g_ffn 7:b_ffn
//  8:W1 9:bb1 10:W2 11:bb2
// Workspace (floats): xn | loga | ssm | fn | h | W1p | W2p  (~93 MB)
// ---------------------------------------------------------------------------
extern "C" void kernel_launch(void* const* d_in, const int* in_sizes, int n_in,
                              void* d_out, int out_size, void* d_ws,
                              size_t ws_size, hipStream_t stream)
{
    (void)in_sizes; (void)n_in; (void)out_size; (void)ws_size;

    const float* x     = (const float*)d_in[0];
    const float* log_A = (const float*)d_in[1];
    const float* w_dt  = (const float*)d_in[2];
    const float* b_dt  = (const float*)d_in[3];
    const float* g_ssm = (const float*)d_in[4];
    const float* b_ssm = (const float*)d_in[5];
    const float* g_ffn = (const float*)d_in[6];
    const float* b_ffn = (const float*)d_in[7];
    const float* W1    = (const float*)d_in[8];
    const float* bb1   = (const float*)d_in[9];
    const float* W2    = (const float*)d_in[10];
    const float* bb2   = (const float*)d_in[11];
    float*       out   = (float*)d_out;

    float* ws   = (float*)d_ws;
    float* xn   = ws;                   // 2,097,152 f
    float* loga = xn + 2097152;         //    65,536 f
    float* ssm  = loga + 65536;         // 2,097,152 f
    float* fn   = ssm + 2097152;        // 2,097,152 f
    float* h    = fn + 2097152;         // 8,388,608 f
    unsigned short* W1p = (unsigned short*)(h + 8388608);   // 8,388,608 ush
    unsigned short* W2p = W1p + 8388608;                    // 8,388,608 ush

    // weight packing (fp32 -> tiled bf16 hi/lo, LDS image layout)
    pack_w_kernel<<<dim3(32, 32), 256, 0, stream>>>(W1, W1p, 1024, 4096);
    pack_w_kernel<<<dim3(8, 128), 256, 0, stream>>>(W2, W2p, 4096, 1024);

    // 1) LN + dt projection
    ln_kernel<true><<<2048, 256, 0, stream>>>(x, g_ssm, b_ssm, w_dt, b_dt,
                                              log_A, xn, loga);
    // 2) SSM recurrence + residual
    ssm_scan_kernel<<<64, 32, 0, stream>>>(xn, loga, x, ssm);
    // 3) FFN LayerNorm
    ln_kernel<false><<<2048, 256, 0, stream>>>(ssm, g_ffn, b_ffn, nullptr,
                                               nullptr, nullptr, fn, nullptr);
    // 4) h = gelu(fn @ W1 + bb1)      M=2048 K=1024 N=4096
    gemm_kernel<0><<<dim3(32, 16), 256, 0, stream>>>(fn, W1p, bb1, nullptr, h,
                                                     2048, 4096, 1024);
    // 5) out = ssm + h @ W2 + bb2     M=2048 K=4096 N=1024
    gemm_kernel<1><<<dim3(8, 16), 256, 0, stream>>>(h, W2p, bb2, ssm, out,
                                                    2048, 1024, 4096);
}
